// CfCAttention_13383118094537
// MI455X (gfx1250) — matrix operand order repaired and checked by reading, not compiled, not run on hardware
//
#include <hip/hip_runtime.h>
#include <hip/hip_bf16.h>
#include <stdint.h>

// ---------------------------------------------------------------------------
// CfC recurrence for MI455X (gfx1250, wave32, WMMA).
//   B=8, T=2048, C=1024.
//   Phase 1: Xf = x @ W_f[:C], Xg = x @ W_g[:C]   (parallel bf16 WMMA GEMMs,
//            2x2 register-tiled for L2-bandwidth efficiency)
//   Phase 2: sequential recurrence, recurrent weights LDS-resident across
//            16 workgroups (256KB/WGP, staged via TDM tensor_load_to_lds),
//            per-step cross-WG h exchange, cluster barrier + atomic barrier.
//   Phase 3: out = y @ W_proj (bf16 WMMA GEMM, f32 out in [B,T,C]).
// ---------------------------------------------------------------------------

typedef __attribute__((ext_vector_type(16))) __bf16    v16bf;
typedef __attribute__((ext_vector_type(8)))  float     v8f;
typedef __attribute__((ext_vector_type(4)))  unsigned  v4u;
typedef __attribute__((ext_vector_type(8)))  int       v8i;
typedef __attribute__((ext_vector_type(4)))  int       v4i;

#define TDIM 2048
#define BDIM 8
#define CDIM 1024
#define MROWS (BDIM * TDIM)   // 16384, row m = t*8 + b  (T-major)
#define KT    (CDIM / 32)     // 32 K-chunks of 32
#define NT    (CDIM / 16)     // 64 N-tiles

__device__ __forceinline__ unsigned short f2bf(float f) {
  unsigned u = __builtin_bit_cast(unsigned, f);
  u += 0x7FFFu + ((u >> 16) & 1u);           // round-to-nearest-even
  return (unsigned short)(u >> 16);
}
__device__ __forceinline__ float bf2f(unsigned short s) {
  unsigned u = ((unsigned)s) << 16;
  return __builtin_bit_cast(float, u);
}

// 32-byte fragment load as two b128 ops.
__device__ __forceinline__ v16bf ldfrag(const unsigned* p) {
  struct S { uint4 a, b; } s;
  s.a = *(const uint4*)(p);
  s.b = *(const uint4*)(p + 4);
  return __builtin_bit_cast(v16bf, s);
}

// ---------------------------------------------------------------------------
// Fragment layouts (dword = 2 bf16, consecutive K):
//  A (16x32, MxK): dword index ((mt*KT+kk)*32 + lane)*8 + v
//     m = mt*16 + lane%16 ; k = kk*32 + (v>=4?16:0) + (lane<16?0:8) + (v&3)*2
//  B (32x16, KxN): dword index ((nt*KT+kk)*32 + lane)*8 + v
//     n = nt*16 + lane%16 ; k = kk*32 + (v>=4?16:0) + (lane<16?0:8) + (v&3)*2
// ---------------------------------------------------------------------------

// Pack the 5 weight blocks (Wf_x, Wf_h, Wg_x, Wg_h, Wp) fp32 -> bf16 B-layout.
__global__ void __launch_bounds__(256) pack_w_kernel(const float* __restrict__ Wf,
                                                     const float* __restrict__ Wg,
                                                     const float* __restrict__ Wp,
                                                     unsigned* __restrict__ dstBase) {
  const int z = blockIdx.z;
  const float* src = (z < 2) ? (Wf + (z == 1 ? (size_t)CDIM * CDIM : 0))
                   : (z < 4) ? (Wg + (z == 3 ? (size_t)CDIM * CDIM : 0))
                             : Wp;
  unsigned* dst = dstBase + (size_t)z * (CDIM * CDIM / 2);
  const int d = blockIdx.x * blockDim.x + threadIdx.x;     // 0..524287
  const int v = d & 7, lane = (d >> 3) & 31, kk = (d >> 8) & 31, nt = d >> 13;
  const int n  = nt * 16 + (lane & 15);
  const int k0 = kk * 32 + ((v >= 4) ? 16 : 0) + ((lane < 16) ? 0 : 8) + (v & 3) * 2;
  const unsigned lo = f2bf(src[(size_t)k0 * CDIM + n]);
  const unsigned hi = f2bf(src[(size_t)(k0 + 1) * CDIM + n]);
  dst[d] = lo | (hi << 16);
}

// Pack x [B,T,C] fp32 -> bf16 A-layout with row m = t*8+b.
__global__ void __launch_bounds__(256) pack_x_kernel(const float* __restrict__ x,
                                                     unsigned* __restrict__ XP) {
  const int d = blockIdx.x * blockDim.x + threadIdx.x;     // 0..8388607
  const int v = d & 7, lane = (d >> 3) & 31, kk = (d >> 8) & 31, mt = d >> 13;
  const int m  = mt * 16 + (lane & 15);
  const int k0 = kk * 32 + ((v >= 4) ? 16 : 0) + ((lane < 16) ? 0 : 8) + (v & 3) * 2;
  const int b = m & 7, t = m >> 3;
  const float* s = x + ((size_t)b * TDIM + t) * CDIM + k0;
  XP[d] = (unsigned)f2bf(s[0]) | ((unsigned)f2bf(s[1]) << 16);
}

// Pack y (bf16 row-major, row m = t*8+b) -> A-layout (dword relabeling only).
__global__ void __launch_bounds__(256) pack_y_kernel(const unsigned* __restrict__ YB,
                                                     unsigned* __restrict__ YP) {
  const int d = blockIdx.x * blockDim.x + threadIdx.x;
  const int v = d & 7, lane = (d >> 3) & 31, kk = (d >> 8) & 31, mt = d >> 13;
  const int m  = mt * 16 + (lane & 15);
  const int k0 = kk * 32 + ((v >= 4) ? 16 : 0) + ((lane < 16) ? 0 : 8) + (v & 3) * 2;
  YP[d] = YB[(size_t)m * (CDIM / 2) + (k0 >> 1)];
}

// Tiled WMMA GEMM: [16384 x 1024] * [1024 x 1024]. 8 waves/block; each wave
// computes a 2x2 grid of 16x16 C-tiles (A/B fragment reuse => 2x arithmetic
// intensity vs 1 tile/wave). Block covers 64 rows x 128 cols.
// outmode 0: bf16 row-major [m][n] (Xf/Xg).  outmode 1: f32 [B,T,C] (d_out).
__global__ void __launch_bounds__(256) gemm_kernel(const unsigned* __restrict__ Apack,
                                                   const unsigned* __restrict__ Bpack,
                                                   void* __restrict__ out, int outmode) {
  const int w = threadIdx.x >> 5, lane = threadIdx.x & 31;
  const int wrow = w >> 2, wcol = w & 3;                  // 2 x 4 wave grid
  const int mt0 = blockIdx.x * 4 + wrow * 2;              // 2 M-tiles per wave
  const int nt0 = blockIdx.y * 8 + wcol * 2;              // 2 N-tiles per wave
  const unsigned* aP = Apack + ((size_t)mt0 * KT * 32 + lane) * 8;
  const unsigned* bP = Bpack + ((size_t)nt0 * KT * 32 + lane) * 8;
  v8f acc00 = {0.f,0.f,0.f,0.f,0.f,0.f,0.f,0.f};
  v8f acc01 = acc00, acc10 = acc00, acc11 = acc00;
  for (int kk = 0; kk < KT; ++kk) {
    v16bf a0 = ldfrag(aP + (size_t)kk * 256);
    v16bf a1 = ldfrag(aP + (size_t)kk * 256 + 8192);      // next M-tile
    v16bf b0 = ldfrag(bP + (size_t)kk * 256);
    v16bf b1 = ldfrag(bP + (size_t)kk * 256 + 8192);      // next N-tile
    acc00 = __builtin_amdgcn_wmma_f32_16x16x32_bf16(false, a0, false, b0, (short)0, acc00, false, false);
    acc01 = __builtin_amdgcn_wmma_f32_16x16x32_bf16(false, a0, false, b1, (short)0, acc01, false, false);
    acc10 = __builtin_amdgcn_wmma_f32_16x16x32_bf16(false, a1, false, b0, (short)0, acc10, false, false);
    acc11 = __builtin_amdgcn_wmma_f32_16x16x32_bf16(false, a1, false, b1, (short)0, acc11, false, false);
  }
  const int rowoff = (lane < 16) ? 0 : 8;
  v8f* accs[4] = {&acc00, &acc01, &acc10, &acc11};
#pragma unroll
  for (int mi = 0; mi < 2; ++mi) {
#pragma unroll
    for (int ni = 0; ni < 2; ++ni) {
      const v8f& a = *accs[mi * 2 + ni];
      const int col = (nt0 + ni) * 16 + (lane & 15);
      if (outmode == 0) {
        unsigned short* o = (unsigned short*)out;
#pragma unroll
        for (int r = 0; r < 8; ++r) {
          const int row = (mt0 + mi) * 16 + r + rowoff;
          o[(size_t)row * CDIM + col] = f2bf(a[r]);
        }
      } else {
        float* o = (float*)out;
#pragma unroll
        for (int r = 0; r < 8; ++r) {
          const int row = (mt0 + mi) * 16 + r + rowoff;   // row = t*8 + b
          const int t = row >> 3, b = row & 7;
          o[((size_t)b * TDIM + t) * CDIM + col] = a[r];
        }
      }
    }
  }
}

// ---------------------------------------------------------------------------
// TDM: async-copy one contiguous 128KB block (8192 x 4 dwords tile) from
// global memory into LDS at byte offset `lds_off`. D# per ISA 08 section 8:
// group0 = {count=1, lds_addr, global_addr[56:0], type=2}; group1 carries
// data_size=4B, tensor_dim0=8192, tensor_dim1=4, tile 8192x4, stride0=8192.
// Tracked by TENSORcnt.
// ---------------------------------------------------------------------------
__device__ __forceinline__ void tdm_load_128k(unsigned lds_off, const void* gsrc) {
  const unsigned long long ga = (unsigned long long)(uintptr_t)gsrc;
  const v4u g0 = { 1u,                                     // count=1, user desc
                   lds_off,                                // lds_addr
                   (unsigned)(ga & 0xFFFFFFFFull),         // global_addr[31:0]
                   (unsigned)((ga >> 32) & 0x1FFFFFFull) | (2u << 30) }; // [56:32] | type=2
  const v8i g1 = { (int)(2u << 16),        // workgroup_mask=0, data_size=4B
                   (int)(8192u << 16),     // tensor_dim0[15:0] @ bits[63:48]
                   (int)(4u << 16),        // dim0[31:16]=0 | tensor_dim1[15:0]=4
                   (int)(8192u << 16),     // dim1[31:16]=0 | tile_dim0=8192
                   (int)4,                 // tile_dim1=4 | tile_dim2=0
                   (int)8192,              // tensor_dim0_stride[31:0]
                   0, 0 };                 // stride0[47:32], stride1
  const v4i gz = {0, 0, 0, 0};
#if __has_include(<hip/amd_detail/amd_gfx1250_TDM.h>)
  const v8i gz8 = {0, 0, 0, 0, 0, 0, 0, 0};
  __builtin_amdgcn_tensor_load_to_lds(g0, g1, gz, gz, gz8, 0);   // clang-23 form
#else
  __builtin_amdgcn_tensor_load_to_lds(g0, g1, gz, gz, 0);        // ROCm 7.2 form
#endif
}

// ---------------------------------------------------------------------------
// Sequential recurrence. 16 workgroups x 256 threads (8 waves). WG `wg` owns
// output columns [wg*64, wg*64+64). LDS: 128KB Wf-slice + 128KB Wg-slice
// (B-frag layout, staged by TDM) + 32KB h A-frag staging + 8KB f32 scratch.
// Wave w: matrix m=w&1 (f/g), local col-tile ct=w>>1. Waves 0..3 do the
// elementwise combine and keep h for their 16 columns register-resident.
// Cross-WG h exchange: bf16 store to YB (also the output y), atomic grid
// barrier + s_cluster_barrier, then scatter YB[t] into A-frag staging.
// ---------------------------------------------------------------------------
__global__ void __launch_bounds__(256) cfc_recurrence_kernel(
    const unsigned* __restrict__ WFHp, const unsigned* __restrict__ WGHp,
    const unsigned short* __restrict__ XF, const unsigned short* __restrict__ XG,
    unsigned short* __restrict__ YB, int* cnt) {
  extern __shared__ char smem[];
  unsigned* WFl = (unsigned*)smem;          // 32768 dwords (4 N-tiles x 32 kk) @ 0
  unsigned* WGl = WFl + 32768;              // 32768 dwords @ 131072
  unsigned* AST = WGl + 32768;              // 8192 dwords: h A-frags, K=1024
  float*    SCR = (float*)(AST + 8192);     // 8 waves x 256 f32

  const int wg = blockIdx.x;                // 0..15
  const int tid = threadIdx.x;
  const int w = tid >> 5, lane = tid & 31;

  // Stage LDS-resident weight slices with the Tensor Data Mover (contiguous
  // 128KB each); zero the h staging (h0 = 0, pad rows 8..15 stay 0 forever).
  if (w == 0) {
    tdm_load_128k(0u,       WFHp + (size_t)wg * 32768);
    tdm_load_128k(131072u,  WGHp + (size_t)wg * 32768);
  }
  for (int i = tid; i < 8192; i += 256) AST[i] = 0u;
  if (w == 0) __builtin_amdgcn_s_wait_tensorcnt(0);
  __syncthreads();

  const int mtx = w & 1, ct = w >> 1;       // matrix select, local col-tile 0..3
  const unsigned* WB = (mtx ? WGl : WFl) + (size_t)ct * 8192 + lane * 8;
  const unsigned* AB = AST + lane * 8;

  float hreg[8];
#pragma unroll
  for (int r = 0; r < 8; ++r) hreg[r] = 0.f;

  for (int t = 0; t < TDIM; ++t) {
    // ---- h_{t-1} @ W_slice : 32 chained WMMAs from LDS ----
    v8f acc = {0.f,0.f,0.f,0.f,0.f,0.f,0.f,0.f};
    for (int kk = 0; kk < 32; ++kk) {
      v16bf a = ldfrag(AB + kk * 256);
      v16bf b = ldfrag(WB + kk * 256);
      acc = __builtin_amdgcn_wmma_f32_16x16x32_bf16(false, a, false, b,
                                                    (short)0, acc, false, false);
    }
#pragma unroll
    for (int r = 0; r < 8; ++r) SCR[w * 256 + r * 32 + lane] = acc[r];
    __syncthreads();

    // ---- combine: f,g, h update (waves 0..3, 16 cols each) ----
    if (w < 4) {
      const int col = wg * 64 + w * 16 + (lane & 15);
#pragma unroll
      for (int r = 0; r < 8; ++r) {
        const size_t rowoff = ((size_t)t * 8 + r) * CDIM + col;
        const float zf = SCR[(2 * w)     * 256 + r * 32 + lane] + bf2f(XF[rowoff]);
        const float zg = SCR[(2 * w + 1) * 256 + r * 32 + lane] + bf2f(XG[rowoff]);
        const float f = 1.f / (1.f + __expf(-zf));
        const float g = tanhf(zg);
        const float hn = f * hreg[r] + (1.f - f) * g;
        hreg[r] = hn;
        if (lane < 16) YB[rowoff] = f2bf(hn);   // h slice -> y (and exchange buf)
        if (t + 1 < TDIM) {                     // global_prefetch next-step rows
          __builtin_prefetch(&XF[rowoff + 8 * CDIM], 0, 1);
          __builtin_prefetch(&XG[rowoff + 8 * CDIM], 0, 1);
        }
      }
    }
    __threadfence();                 // release h slice to other WGPs (via L2)
    __syncthreads();

    // ---- all-WG sync: atomic counter barrier + CDNA5 cluster barrier ----
    if (tid == 0) {
      atomicAdd(cnt, 1);
      volatile int* vc = cnt;
      while (*vc < 16 * (t + 1)) __builtin_amdgcn_s_sleep(2);
    }
    __builtin_amdgcn_s_cluster_barrier();  // s_barrier_signal/wait -3 (NOP if unclustered)
    __syncthreads();
    __threadfence();                 // acquire

    // ---- gather full h_t (8x1024 bf16, 16KB from L2) into A-frag staging ----
    const unsigned* YBd = (const unsigned*)YB + (size_t)t * 4096;  // 8 rows x 512 dw
    for (int i = tid; i < 4096; i += 256) {
      const int b = i >> 9, j = i & 511;          // j = k/2
      const int k0 = j * 2, kl = k0 & 31, sub = kl & 15;
      const int lt = (sub < 8) ? b : b + 16;
      const int vv = (kl >> 4) * 4 + ((sub & 7) >> 1);
      AST[(k0 >> 5) * 256 + lt * 8 + vv] = YBd[b * 512 + j];
    }
    __syncthreads();
  }
}

// ---------------------------------------------------------------------------
extern "C" void kernel_launch(void* const* d_in, const int* in_sizes, int n_in,
                              void* d_out, int out_size, void* d_ws, size_t ws_size,
                              hipStream_t stream) {
  const float* x  = (const float*)d_in[0];
  const float* Wf = (const float*)d_in[1];   // [2C, C]
  const float* Wg = (const float*)d_in[2];   // [2C, C]
  const float* Wp = (const float*)d_in[3];   // [C, C]

  char* ws = (char*)d_ws;
  size_t o = 0;
  int* cnt = (int*)(ws + o);                      o += 256;
  unsigned* WPK = (unsigned*)(ws + o);            o += 5ull * 2 * 1024 * 1024;  // 5 packed weights
  unsigned* XP  = (unsigned*)(ws + o);            o += 32ull * 1024 * 1024;     // packed x, reused for packed y
  unsigned short* XF = (unsigned short*)(ws + o); o += 32ull * 1024 * 1024;
  unsigned short* XG = (unsigned short*)(ws + o); o += 32ull * 1024 * 1024;
  unsigned short* YB = (unsigned short*)(ws + o); o += 32ull * 1024 * 1024;     // y bf16, row m=t*8+b

  unsigned* WFX = WPK + 0 * 524288;   // W_f[:C]   B-layout
  unsigned* WFH = WPK + 1 * 524288;   // W_f[C:]   B-layout (recurrent)
  unsigned* WGX = WPK + 2 * 524288;
  unsigned* WGH = WPK + 3 * 524288;
  unsigned* WPp = WPK + 4 * 524288;

  hipMemsetAsync(cnt, 0, 4, stream);              // reset grid barrier (graph-safe)

  const dim3 blk(256);
  pack_w_kernel<<<dim3(2048, 1, 5), blk, 0, stream>>>(Wf, Wg, Wp, WPK);
  pack_x_kernel<<<dim3(32768), blk, 0, stream>>>(x, XP);

  // Parallel input projections: Xf = x@Wf[:C], Xg = x@Wg[:C]
  gemm_kernel<<<dim3(256, 8), blk, 0, stream>>>(XP, WFX, (void*)XF, 0);
  gemm_kernel<<<dim3(256, 8), blk, 0, stream>>>(XP, WGX, (void*)XG, 0);

  // Sequential recurrence: 16 WGs, 296KB dynamic LDS each
  hipFuncSetAttribute((const void*)cfc_recurrence_kernel,
                      hipFuncAttributeMaxDynamicSharedMemorySize, 303104);
  cfc_recurrence_kernel<<<dim3(16), blk, 303104, stream>>>(WFH, WGH, XF, XG, YB, cnt);

  // Final projection: out = y @ W_proj, written as [B,T,C] fp32
  pack_y_kernel<<<dim3(32768), blk, 0, stream>>>((const unsigned*)YB, XP);
  gemm_kernel<<<dim3(256, 8), blk, 0, stream>>>(XP, WPp, d_out, 1);
}
